// PointPositionEmbedding_76656576299160
// MI455X (gfx1250) — matrix operand order repaired and checked by reading, not compiled, not run on hardware
//
#include <hip/hip_runtime.h>

typedef __attribute__((ext_vector_type(16))) _Float16 v16h;
typedef __attribute__((ext_vector_type(8)))  _Float16 v8h;
typedef __attribute__((ext_vector_type(8)))  float    v8f;
typedef __attribute__((ext_vector_type(4)))  float    v4f;
typedef __attribute__((ext_vector_type(4)))  unsigned v4u;

#define N_PTS 4096
#define TILES_PER_WAVE 8
#define WAVES_PER_BLOCK 8

// Transposed formulation: hT = W1^T x feat^T, outT = W2^T x hT.
// GEMM1 uses V_WMMA_F16_16X16X32_F16 (f16 C/D): hT comes out f16-packed with
// 8 consecutive dims per lane; ReLU = raw inline-asm v_pk_max_num_f16 (no
// canonicalize); staging = 1 ds_store_b128 per 16-dim tile. GEMM2 keeps f32
// accumulation + f32 bias. Global accesses use uniform-base + 32-bit offsets
// (GVS addressing) so per-tile address updates are single 32-bit adds.
__global__ __launch_bounds__(256)
void ppe_wmma_kernel(const float* __restrict__ xyz,
                     const float* __restrict__ dist,
                     const float* __restrict__ W1,
                     const float* __restrict__ b1,
                     const float* __restrict__ W2,
                     const float* __restrict__ b2,
                     const long long* __restrict__ idx,
                     float* __restrict__ out)
{
    // Weights staged transposed: wt[dim][k] = W[k][dim], f16. W1 K padded 10->32.
    __shared__ __align__(32) _Float16 w1t[64][32];
    __shared__ __align__(32) _Float16 w2t[64][64];
    __shared__ __align__(16) _Float16 b1s[64];   // b1 in f16 (loads, not remat-able cvts)
    // Per-wave h tile, row-major [row][hidden], stride 72 halves (144B rows, 16B aligned).
    __shared__ __align__(16) _Float16 hst[WAVES_PER_BLOCK][16][72];

    const int tid = threadIdx.x;

    for (int i = tid; i < 64 * 32; i += 256) {
        const int n = i >> 5, k = i & 31;
        w1t[n][k] = (k < 10) ? (_Float16)W1[k * 64 + n] : (_Float16)0.0f;
    }
    for (int i = tid; i < 64 * 64; i += 256) {
        const int n = i >> 6, k = i & 63;
        w2t[n][k] = (_Float16)W2[k * 64 + n];
    }
    if (tid < 64) b1s[tid] = (_Float16)b1[tid];
    __syncthreads();

    const int wave = tid >> 5;
    const int lane = tid & 31;
    const int m    = lane & 15;   // A: dim row within tile | B/C: point-row column
    const int half = lane >> 4;
    const int ko   = half * 8;

    // Hoisted A fragments (weights) in contiguous k order (k = 16*half + j):
    // lane holds row m, elements {2*ko..2*ko+15} (+K-chunk offset).
    v16h w1a[4];
    v16h w2a[4][2];
#pragma unroll
    for (int nb = 0; nb < 4; ++nb) {
        const int n = nb * 16 + m;
        {
            const v8h lo = *(const v8h*)&w1t[n][2 * ko];
            const v8h hi = *(const v8h*)&w1t[n][2 * ko + 8];
#pragma unroll
            for (int j = 0; j < 8; ++j) { w1a[nb][j] = lo[j]; w1a[nb][j + 8] = hi[j]; }
        }
#pragma unroll
        for (int kc = 0; kc < 2; ++kc) {
            const v8h lo = *(const v8h*)&w2t[n][kc * 32 + 2 * ko];
            const v8h hi = *(const v8h*)&w2t[n][kc * 32 + 2 * ko + 8];
#pragma unroll
            for (int j = 0; j < 8; ++j) { w2a[nb][kc][j] = lo[j]; w2a[nb][kc][j + 8] = hi[j]; }
        }
    }

    // Bias as C-init: element q corresponds to dim nb*16 + 8*half + q.
    v8h b1h[4];
    v8f b2c[4];
#pragma unroll
    for (int nb = 0; nb < 4; ++nb) {
        b1h[nb] = *(const v8h*)&b1s[nb * 16 + 8 * half];
        b2c[nb] = *(const v8f*)(b2 + nb * 16 + 8 * half);
    }

    const long long wave_id = (long long)blockIdx.x * WAVES_PER_BLOCK + wave;
    const long long tile0   = wave_id * TILES_PER_WAVE;

    // Loop-invariant LDS pointers (lane-dependent only).
    _Float16*       hstore = &hst[wave][m][8 * half];     // + nb*16 per tile-column
    const _Float16* hload0 = &hst[wave][m][half * 16];    // K 0..31 fragment
    const _Float16* hload1 = hload0 + 32;                 // K 32..63 fragment

    // Wave-constant neighbor-gather base: a wave's 8 tiles share one batch
    // (tiles are 8-aligned and 4096 % 8 == 0).
    const float* gbase = xyz + (long long)(tile0 >> 12) * (N_PTS * 3);
    // idx low dwords: gather indices are < 4096, so only the low half matters.
    const int* idx32 = (const int*)idx;

    // 32-bit offsets against uniform 64-bit bases (GVS addressing).
    const unsigned coff = (unsigned)(tile0 * 3);                       // xyz (uniform)
    unsigned       roff = (unsigned)(tile0 * 16) + m;                  // idx/dist rows
    unsigned       ooff = (unsigned)(tile0 * 1024) + m * 64 + 8 * half;// out floats

    for (int t = 0; t < TILES_PER_WAVE; ++t) {
        // Wave-uniform center point -> scalar loads with immediate offsets.
        const float cx = xyz[coff + 3 * t + 0];
        const float cy = xyz[coff + 3 * t + 1];
        const float cz = xyz[coff + 3 * t + 2];

        // feat^T as B operand: lane = point-row m, elements j <-> K = half*16 + j.
        // Only K=0..9 are nonzero -> only lanes 0-15 gather.
        v16h fB = {};
        if (half == 0) {
            const int jni = idx32[roff * 2];            // low dword of int64 index
            const float* gp = gbase + jni * 3;
            const float nx = gp[0], ny = gp[1], nz = gp[2];   // the only true scatter
            const float dd = dist[roff];
            fB[0] = (_Float16)cx;        fB[1] = (_Float16)cy;
            fB[2] = (_Float16)cz;        fB[3] = (_Float16)nx;
            fB[4] = (_Float16)ny;        fB[5] = (_Float16)nz;
            fB[6] = (_Float16)(cx - nx); fB[7] = (_Float16)(cy - ny);
            fB[8] = (_Float16)(cz - nz); fB[9] = (_Float16)dd;
        }

        // GEMM1 (f16 C/D): hT tile per nb; ReLU = 4 raw v_pk_max_num_f16; 16B LDS store.
#pragma unroll
        for (int nb = 0; nb < 4; ++nb) {
            v8h hh = __builtin_amdgcn_wmma_f16_16x16x32_f16(
                         false, w1a[nb], false, fB, (short)0, b1h[nb], false, false);
            v4u u;
            __builtin_memcpy(&u, &hh, 16);
            unsigned a0 = u[0], a1 = u[1], a2 = u[2], a3 = u[3];
            asm("v_pk_max_num_f16 %0, %0, 0" : "+v"(a0));
            asm("v_pk_max_num_f16 %0, %0, 0" : "+v"(a1));
            asm("v_pk_max_num_f16 %0, %0, 0" : "+v"(a2));
            asm("v_pk_max_num_f16 %0, %0, 0" : "+v"(a3));
            const v4u r = {a0, a1, a2, a3};
            // lane's row = m; dims nb*16 + 8*half + 0..7 are contiguous, f16-packed.
            *(v4u*)(hstore + nb * 16) = r;
        }

        // Compiler memory fence + DS drain: cross-lane LDS exchange below.
        asm volatile("s_wait_dscnt 0" ::: "memory");

        // h^T as B operand: lane = row m, element j <-> k = kc*32 + half*16 + j.
        const v16h hB0 = *(const v16h*)hload0;
        const v16h hB1 = *(const v16h*)hload1;

#pragma unroll
        for (int nb = 0; nb < 4; ++nb) {
            v8f c8 = b2c[nb];
            c8 = __builtin_amdgcn_wmma_f32_16x16x32_f16(
                     false, w2a[nb][0], false, hB0, (short)0, c8, false, false);
            c8 = __builtin_amdgcn_wmma_f32_16x16x32_f16(
                     false, w2a[nb][1], false, hB1, (short)0, c8, false, false);
            // lane's row = m; dims nb*16 + 8*half + 0..7 -> two b128 stores.
            float* p = out + ooff + nb * 16;
            *(v4f*)p       = __builtin_shufflevector(c8, c8, 0, 1, 2, 3);
            *(v4f*)(p + 4) = __builtin_shufflevector(c8, c8, 4, 5, 6, 7);
        }

        roff += 16;
        ooff += 1024;
    }
}

extern "C" void kernel_launch(void* const* d_in, const int* in_sizes, int n_in,
                              void* d_out, int out_size, void* d_ws, size_t ws_size,
                              hipStream_t stream) {
    const float*     xyz  = (const float*)d_in[0];
    const float*     dist = (const float*)d_in[1];
    const float*     W1   = (const float*)d_in[2];
    const float*     b1   = (const float*)d_in[3];
    const float*     W2   = (const float*)d_in[4];
    const float*     b2   = (const float*)d_in[5];
    const long long* idx  = (const long long*)d_in[6];
    float* out = (float*)d_out;

    // 16*4096*16 rows -> 65536 16-row tiles -> 8192 waves -> 1024 blocks.
    const int total_tiles = (16 * 4096 * 16) / 16;
    const int blocks = total_tiles / (TILES_PER_WAVE * WAVES_PER_BLOCK);
    ppe_wmma_kernel<<<blocks, 256, 0, stream>>>(xyz, dist, W1, b1, W2, b2, idx, out);
}
